// GraphDecoder_9509057593395
// MI455X (gfx1250) — matrix operand order repaired and checked by reading, not compiled, not run on hardware
//
#include <hip/hip_runtime.h>
#include <hip/hip_bf16.h>
#include <math.h>

// Problem constants
#define BATCH   128
#define LATENT  256
#define HIDDEN  512
#define H2DIM   256
#define NF      64
#define MAXN    50
#define NODED   (MAXN * NF)          // 3200
#define PAIRS   (MAXN * (MAXN - 1))  // 2450
#define EDGE_TILE 64                 // pairs per block (4 x 16-row WMMA tiles)
#define EPB     ((PAIRS + EDGE_TILE - 1) / EDGE_TILE)  // 39 tiles per batch row

typedef __bf16 bf16;
typedef __attribute__((ext_vector_type(16))) __bf16 v16bf;
typedef __attribute__((ext_vector_type(8)))  __bf16 v8bf;
typedef __attribute__((ext_vector_type(8)))  float  v8f;

union V16U { v16bf v; v8bf h[2]; };

// ---------------------------------------------------------------------------
// WMMA fragment loaders (ISA 7.12.2 wave32 layouts)
// A (16xK, row-major, K contiguous): lane L row=L&15; K-halves {kb..kb+7, kb+16..kb+23}, kb=(L&16)?8:0
__device__ __forceinline__ v16bf load_frag_a_rm(const bf16* __restrict__ A, int lda, int lane, int k0) {
    int r  = lane & 15;
    int kb = (lane & 16) ? 8 : 0;
    const bf16* p = A + (size_t)r * lda + k0 + kb;
    V16U u;
    u.h[0] = *(const v8bf*)(p);
    u.h[1] = *(const v8bf*)(p + 16);
    return u.v;
}
// B = W^T tile (B[k][n] = W[n][k], W row-major [N,K]): lane L col=L&15; K-halves {kb..kb+7, kb+16..kb+23}, kb=(L&16)?8:0
__device__ __forceinline__ v16bf load_frag_b_wt(const bf16* __restrict__ W, int ldb, int lane, int k0) {
    int n  = lane & 15;
    int kb = (lane & 16) ? 8 : 0;
    const bf16* p = W + (size_t)n * ldb + k0 + kb;
    V16U u;
    u.h[0] = *(const v8bf*)(p);
    u.h[1] = *(const v8bf*)(p + 16);
    return u.v;
}

// ---------------------------------------------------------------------------
__global__ void f32_to_bf16_kernel(const float* __restrict__ src, bf16* __restrict__ dst, int n) {
    int i = blockIdx.x * blockDim.x + threadIdx.x;
    int stride = gridDim.x * blockDim.x;
    for (; i < n; i += stride) dst[i] = (bf16)src[i];
}

// ---------------------------------------------------------------------------
// Generic C = act(A @ W^T + bias); one 16x16 tile per wave, bf16 WMMA, f32 accum.
enum { EPI_RELU_BF16 = 0, EPI_SIG_F32_BF16 = 1, EPI_F32 = 2 };

template <int MODE>
__global__ __launch_bounds__(256)
void wmma_gemm_wt(const bf16* __restrict__ A, int lda,
                  const bf16* __restrict__ W, int ldb,
                  const float* __restrict__ bias,
                  float* __restrict__ outF, bf16* __restrict__ outB,
                  int ldc, int M, int N, int K)
{
    int wave = (blockIdx.x * blockDim.x + threadIdx.x) >> 5;
    int lane = threadIdx.x & 31;
    int tilesN = N >> 4;
    int totalTiles = (M >> 4) * tilesN;
    if (wave >= totalTiles) return;                 // wave-uniform: EXEC stays all-ones
    int tm = wave / tilesN, tn = wave % tilesN;

    const bf16* Ab = A + (size_t)(tm * 16) * lda;
    const bf16* Wb = W + (size_t)(tn * 16) * ldb;

    v8f acc = {};
    for (int k0 = 0; k0 < K; k0 += 32) {
        v16bf a = load_frag_a_rm(Ab, lda, lane, k0);
        v16bf b = load_frag_b_wt(Wb, ldb, lane, k0);
        acc = __builtin_amdgcn_wmma_f32_16x16x32_bf16(false, a, false, b, (short)0, acc, false, false);
    }

    int n = tn * 16 + (lane & 15);
    int mbase = tm * 16 + ((lane & 16) ? 8 : 0);
    float bv = bias ? bias[n] : 0.0f;
#pragma unroll
    for (int g = 0; g < 8; ++g) {
        float v = acc[g] + bv;
        size_t idx = (size_t)(mbase + g) * ldc + n;
        if (MODE == EPI_RELU_BF16) {
            outB[idx] = (bf16)fmaxf(v, 0.0f);
        } else if (MODE == EPI_SIG_F32_BF16) {
            outF[idx] = 1.0f / (1.0f + __expf(-v));  // node_probs
            outB[idx] = (bf16)v;                     // raw node_feat for edge path
        } else {
            outF[idx] = v;
        }
    }
}

// ---------------------------------------------------------------------------
// Fused edge kernel: per block = one (b, 64-pair) tile = 4 WMMA row-tiles.
//  Phase 1: e1 = relu(Zc[b] + Ai[b,i] + Aj[b,j]) -> bf16 LDS tile [64,512]  (Zc holds +c1)
//  Phase 2: 16 waves x 4 row-tiles: e2 = relu(e1 @ E2^T + c2); one B-frag feeds 4 WMMAs
//           (4x arithmetic intensity vs one-tile blocks: ~64 FLOP per L2 byte of E2)
//  Phase 3: edge_prob = sigmoid(e2 . E3 + c3), 8 lanes per row + shuffle reduce
__global__ __launch_bounds__(512)
void edge_kernel(const float* __restrict__ Zc,    // [B][512]   (includes c1)
                 const float* __restrict__ Ai,    // [B*50][512]
                 const float* __restrict__ Aj,    // [B*50][512]
                 const bf16*  __restrict__ E2b,   // [256][512] bf16 row-major
                 const float* __restrict__ c2,    // [256]
                 const float* __restrict__ E3,    // [256]
                 const float* __restrict__ c3,    // [1]
                 float* __restrict__ outE)        // [B][2450]
{
    __shared__ __align__(16) bf16  e1s[EDGE_TILE * HIDDEN];  // 64 KB
    __shared__ __align__(16) float e2s[EDGE_TILE * H2DIM];   // 64 KB
    __shared__ int nodeI[EDGE_TILE], nodeJ[EDGE_TILE];

    int b   = blockIdx.x / EPB;
    int pt  = blockIdx.x % EPB;
    int p0  = pt * EDGE_TILE;
    int tid = threadIdx.x;

    if (tid < EDGE_TILE) {
        int p = p0 + tid; if (p >= PAIRS) p = PAIRS - 1;   // clamp tail rows
        int i = p / (MAXN - 1);
        int r = p % (MAXN - 1);
        nodeI[tid] = i;
        nodeJ[tid] = r + (r >= i ? 1 : 0);
    }
    __syncthreads();

    // Phase 1: build e1 tile; each thread: 64 contiguous cols of one row
    {
        int row = tid >> 3;            // 0..63
        int c0  = (tid & 7) * 64;      // 0,64,...,448
        const float* zr = Zc + (size_t)b * HIDDEN;
        const float* ar = Ai + ((size_t)b * MAXN + nodeI[row]) * HIDDEN;
        const float* br = Aj + ((size_t)b * MAXN + nodeJ[row]) * HIDDEN;
#pragma unroll 8
        for (int c = c0; c < c0 + 64; ++c) {
            float v = zr[c] + ar[c] + br[c];
            e1s[row * HIDDEN + c] = (bf16)fmaxf(v, 0.0f);
        }
    }
    __syncthreads();

    // Phase 2: wave w -> output cols [w*16, w*16+16) for all 4 row-tiles, K = 512
    {
        int wave = tid >> 5;
        int lane = tid & 31;
        const bf16* Wb = E2b + (size_t)(wave * 16) * HIDDEN;
        v8f acc0 = {}, acc1 = {}, acc2 = {}, acc3 = {};
        for (int k0 = 0; k0 < HIDDEN; k0 += 32) {
            v16bf bb = load_frag_b_wt(Wb, HIDDEN, lane, k0);            // one L2 read...
            v16bf a0 = load_frag_a_rm(e1s + 0 * 16 * HIDDEN, HIDDEN, lane, k0);
            v16bf a1 = load_frag_a_rm(e1s + 1 * 16 * HIDDEN, HIDDEN, lane, k0);
            v16bf a2 = load_frag_a_rm(e1s + 2 * 16 * HIDDEN, HIDDEN, lane, k0);
            v16bf a3 = load_frag_a_rm(e1s + 3 * 16 * HIDDEN, HIDDEN, lane, k0);
            acc0 = __builtin_amdgcn_wmma_f32_16x16x32_bf16(false, a0, false, bb, (short)0, acc0, false, false);
            acc1 = __builtin_amdgcn_wmma_f32_16x16x32_bf16(false, a1, false, bb, (short)0, acc1, false, false);
            acc2 = __builtin_amdgcn_wmma_f32_16x16x32_bf16(false, a2, false, bb, (short)0, acc2, false, false);
            acc3 = __builtin_amdgcn_wmma_f32_16x16x32_bf16(false, a3, false, bb, (short)0, acc3, false, false);
        }
        int n  = wave * 16 + (lane & 15);
        int mb = (lane & 16) ? 8 : 0;
        float cb = c2[n];
#pragma unroll
        for (int g = 0; g < 8; ++g) {
            e2s[( 0 + mb + g) * H2DIM + n] = fmaxf(acc0[g] + cb, 0.0f);
            e2s[(16 + mb + g) * H2DIM + n] = fmaxf(acc1[g] + cb, 0.0f);
            e2s[(32 + mb + g) * H2DIM + n] = fmaxf(acc2[g] + cb, 0.0f);
            e2s[(48 + mb + g) * H2DIM + n] = fmaxf(acc3[g] + cb, 0.0f);
        }
    }
    __syncthreads();

    // Phase 3: 8 threads per row, 32 cols each, shuffle-reduce within width-8 groups
    {
        int row = tid >> 3;            // 0..63
        int seg = tid & 7;
        const float* er = e2s + row * H2DIM + seg * 32;
        const float* wr = E3 + seg * 32;
        float s = 0.0f;
#pragma unroll 8
        for (int g = 0; g < 32; ++g) s += er[g] * wr[g];
        s += __shfl_down(s, 4, 8);
        s += __shfl_down(s, 2, 8);
        s += __shfl_down(s, 1, 8);
        if (seg == 0) {
            int p = p0 + row;
            if (p < PAIRS) {
                float v = s + c3[0];
                outE[(size_t)b * PAIRS + p] = 1.0f / (1.0f + __expf(-v));
            }
        }
    }
}

// ---------------------------------------------------------------------------
extern "C" void kernel_launch(void* const* d_in, const int* in_sizes, int n_in,
                              void* d_out, int out_size, void* d_ws, size_t ws_size,
                              hipStream_t stream) {
    const float* z  = (const float*)d_in[0];
    const float* W1 = (const float*)d_in[1];
    const float* b1 = (const float*)d_in[2];
    const float* W2 = (const float*)d_in[3];
    const float* b2 = (const float*)d_in[4];
    const float* W3 = (const float*)d_in[5];
    const float* b3 = (const float*)d_in[6];
    const float* E1 = (const float*)d_in[7];
    const float* c1 = (const float*)d_in[8];
    const float* E2 = (const float*)d_in[9];
    const float* c2 = (const float*)d_in[10];
    const float* E3 = (const float*)d_in[11];
    const float* c3 = (const float*)d_in[12];

    float* outNode = (float*)d_out;                         // [128][3200]
    float* outEdge = (float*)d_out + (size_t)BATCH * NODED; // [128][2450]

    char* ws = (char*)d_ws;
    size_t off = 0;
    auto alloc = [&](size_t bytes) -> void* {
        void* p = ws + off;
        off = (off + bytes + 255) & ~(size_t)255;
        return p;
    };
    bf16*  zb  = (bf16*)alloc((size_t)BATCH  * LATENT * 2);
    bf16*  W1b = (bf16*)alloc((size_t)HIDDEN * LATENT * 2);
    bf16*  W2b = (bf16*)alloc((size_t)HIDDEN * HIDDEN * 2);
    bf16*  W3b = (bf16*)alloc((size_t)NODED  * HIDDEN * 2);
    bf16*  E1b = (bf16*)alloc((size_t)HIDDEN * (LATENT + 2 * NF) * 2);
    bf16*  E2b = (bf16*)alloc((size_t)H2DIM  * HIDDEN * 2);
    bf16*  h1b = (bf16*)alloc((size_t)BATCH  * HIDDEN * 2);
    bf16*  h2b = (bf16*)alloc((size_t)BATCH  * HIDDEN * 2);
    bf16*  nfb = (bf16*)alloc((size_t)BATCH  * NODED * 2);
    float* Zc  = (float*)alloc((size_t)BATCH * HIDDEN * 4);
    float* Ai  = (float*)alloc((size_t)BATCH * MAXN * HIDDEN * 4);
    float* Aj  = (float*)alloc((size_t)BATCH * MAXN * HIDDEN * 4);

    auto cvt = [&](const float* s, bf16* d, int n) {
        int blk = (n + 255) / 256;
        f32_to_bf16_kernel<<<blk, 256, 0, stream>>>(s, d, n);
    };
    cvt(z,  zb,  BATCH * LATENT);
    cvt(W1, W1b, HIDDEN * LATENT);
    cvt(W2, W2b, HIDDEN * HIDDEN);
    cvt(W3, W3b, NODED * HIDDEN);
    cvt(E1, E1b, HIDDEN * (LATENT + 2 * NF));
    cvt(E2, E2b, H2DIM * HIDDEN);

    const int E1LD = LATENT + 2 * NF; // 384
    auto blocks = [](int tiles) { return (tiles + 7) / 8; };

    // h1 = relu(z @ W1^T + b1)
    wmma_gemm_wt<EPI_RELU_BF16><<<blocks((BATCH/16)*(HIDDEN/16)), 256, 0, stream>>>(
        zb, LATENT, W1b, LATENT, b1, nullptr, h1b, HIDDEN, BATCH, HIDDEN, LATENT);
    // h2 = relu(h1 @ W2^T + b2)
    wmma_gemm_wt<EPI_RELU_BF16><<<blocks((BATCH/16)*(HIDDEN/16)), 256, 0, stream>>>(
        h1b, HIDDEN, W2b, HIDDEN, b2, nullptr, h2b, HIDDEN, BATCH, HIDDEN, HIDDEN);
    // node_feat = h2 @ W3^T + b3 ; node_probs = sigmoid -> d_out ; bf16 stash -> nfb
    wmma_gemm_wt<EPI_SIG_F32_BF16><<<blocks((BATCH/16)*(NODED/16)), 256, 0, stream>>>(
        h2b, HIDDEN, W3b, HIDDEN, b3, outNode, nfb, NODED, BATCH, NODED, HIDDEN);
    // Zc = z @ E1z^T + c1   (E1 cols 0..255)
    wmma_gemm_wt<EPI_F32><<<blocks((BATCH/16)*(HIDDEN/16)), 256, 0, stream>>>(
        zb, LATENT, E1b, E1LD, c1, Zc, nullptr, HIDDEN, BATCH, HIDDEN, LATENT);
    // Ai = node_feat @ E1i^T  (E1 cols 256..319), rows = B*MAXN = 6400
    wmma_gemm_wt<EPI_F32><<<blocks((BATCH*MAXN/16)*(HIDDEN/16)), 256, 0, stream>>>(
        nfb, NF, E1b + LATENT, E1LD, nullptr, Ai, nullptr, HIDDEN, BATCH*MAXN, HIDDEN, NF);
    // Aj = node_feat @ E1j^T  (E1 cols 320..383)
    wmma_gemm_wt<EPI_F32><<<blocks((BATCH*MAXN/16)*(HIDDEN/16)), 256, 0, stream>>>(
        nfb, NF, E1b + LATENT + NF, E1LD, nullptr, Aj, nullptr, HIDDEN, BATCH*MAXN, HIDDEN, NF);

    // Fused edge MLP (~82 GFLOP of WMMA), 64 pairs / block, B-frag reuse x4
    edge_kernel<<<BATCH * EPB, 512, 0, stream>>>(Zc, Ai, Aj, E2b, c2, E3, c3, outEdge);
}
